// T5block0_36498632081683
// MI455X (gfx1250) — matrix operand order repaired and checked by reading, not compiled, not run on hardware
//
#include <hip/hip_runtime.h>

// ---------------------------------------------------------------------------
// T5 block (B=1, S=512, D=4096, H=64, DK=64, F=10240) for MI455X (gfx1250).
//  * GEMMs: f32 weights stream from HBM (~850MB once; 23.3TB/s => ~37us floor),
//    converted to bf16 via native fptrunc (v_cvt_pk_bf16_f32) in a
//    double-buffered global->register->LDS pipeline, computed with
//    v_wmma_f32_16x16x32_bf16 (f32 accum). QKV fused (z=3), wi0/wi1 fused (z=2).
//  * Attention (~2% of FLOPs): k/v head slices staged into LDS by the Tensor
//    Data Mover (tensor_load_to_lds, D# built per ISA 8.3/8.4; TDM pad gives
//    k a 68-float row stride => conflict-free score loop), f32 softmax.
// ---------------------------------------------------------------------------

#define S_LEN 512
#define D_DIM 4096
#define H_NUM 64
#define DK_DIM 64
#define F_DIM 10240
#define EPSV 1e-6f

typedef __attribute__((ext_vector_type(16))) __bf16 v16bf;
typedef __attribute__((ext_vector_type(8)))  __bf16 v8bf;
typedef __attribute__((ext_vector_type(4)))  __bf16 v4bf;
typedef __attribute__((ext_vector_type(8)))  float  v8f;
typedef __attribute__((ext_vector_type(4)))  float  v4f;
typedef unsigned int u32x4 __attribute__((ext_vector_type(4)));
typedef int          i32x4 __attribute__((ext_vector_type(4)));
typedef int          i32x8 __attribute__((ext_vector_type(8)));

#if defined(__HIP_DEVICE_COMPILE__) && __has_builtin(__builtin_amdgcn_tensor_load_to_lds) && __has_builtin(__builtin_amdgcn_s_wait_tensorcnt)
#define HAVE_TDM 1
#else
#define HAVE_TDM 0
#endif

// 4xf32 -> 4xbf16 RNE; fptrunc vector form selects packed v_cvt_pk_bf16_f32
static __device__ __forceinline__ v4bf cvt4(v4f v) {
  return __builtin_convertvector(v, v4bf);
}

// ---------------------------------------------------------------------------
// RMSNorm (T5 style: no mean subtraction, no bias)
// ---------------------------------------------------------------------------
__global__ __launch_bounds__(256) void rmsnorm_f32_kernel(
    const float* __restrict__ x, const float* __restrict__ w,
    float* __restrict__ out) {
  __shared__ float red[256];
  const int row = blockIdx.x;
  const int tid = threadIdx.x;
  const float* xr = x + (size_t)row * D_DIM;
  float ss = 0.f;
  for (int i = tid; i < D_DIM; i += 256) { float v = xr[i]; ss += v * v; }
  red[tid] = ss;
  __syncthreads();
  for (int off = 128; off > 0; off >>= 1) {
    if (tid < off) red[tid] += red[tid + off];
    __syncthreads();
  }
  const float rs = rsqrtf(red[0] * (1.0f / (float)D_DIM) + EPSV);
  float* orow = out + (size_t)row * D_DIM;
  for (int i = tid; i < D_DIM; i += 256) orow[i] = xr[i] * rs * w[i];
}

// ---------------------------------------------------------------------------
// GEMM: C[M,N] = A[M,K] * Bz[K,N] (+ Resid); Bz/Cz selected by blockIdx.z so
// independent GEMMs sharing A (QKV, wi0/wi1) fuse into one launch.
// Block tile 128x128, BK=32, 256 threads = 8 wave32s (2Mx4N), wave = 64x32 =
// 4x2 WMMA tiles. Double-buffered LDS: issue next-tile global loads, run
// WMMAs on current buffer, convert/store into the other buffer, one barrier.
// Requires M%128==0, N%128==0, K%32==0 (true for all uses here).
// ---------------------------------------------------------------------------
#define BM 128
#define BN 128
#define BK 32

__global__ __launch_bounds__(256) void gemm_bf16_wmma_kernel(
    const float* __restrict__ A,
    const float* __restrict__ B0, const float* __restrict__ B1,
    const float* __restrict__ B2,
    const float* __restrict__ Resid,
    float* __restrict__ C0, float* __restrict__ C1, float* __restrict__ C2,
    int M, int N, int K) {
  __shared__ __align__(64) __bf16 As[2][BM][BK];   // row-major (m, k)
  __shared__ __align__(64) __bf16 Bs[2][BN][BK];   // transposed (n, k)

  const float* B = (blockIdx.z == 0) ? B0 : ((blockIdx.z == 1) ? B1 : B2);
  float*       C = (blockIdx.z == 0) ? C0 : ((blockIdx.z == 1) ? C1 : C2);

  const int tid  = threadIdx.x;
  const int lane = tid & 31;
  const int wave = tid >> 5;
  const int wm   = wave >> 2;        // 0..1  (64-row strip)
  const int wn   = wave & 3;         // 0..3  (32-col strip)
  const int row0 = blockIdx.x * BM;
  const int col0 = blockIdx.y * BN;
  const int m16  = lane & 15;
  const int sel  = lane >> 4;        // 0/1: which K-half this lane holds
  const int koff = sel * 8;

  // per-thread staging slots (4 float4 of A, 4 float4 of B per tile)
  int aRow[4], aK[4], bKK[4], bN4[4];
#pragma unroll
  for (int g = 0; g < 4; ++g) {
    int slot = tid + g * 256;
    aRow[g] = slot >> 3;            // 0..127
    aK[g]   = (slot & 7) * 4;       // 0..28
    bKK[g]  = slot >> 5;            // 0..31
    bN4[g]  = (slot & 31) * 4;      // 0..124
  }

  v4f aR[4], bR[4];
  auto load_regs = [&](int kbase) {
#pragma unroll
    for (int g = 0; g < 4; ++g) {
      aR[g] = *(const v4f*)(A + (size_t)(row0 + aRow[g]) * K + kbase + aK[g]);
      bR[g] = *(const v4f*)(B + (size_t)(kbase + bKK[g]) * N + col0 + bN4[g]);
    }
  };
  auto commit = [&](int buf) {  // f32 -> bf16 (packed cvt) + LDS store
#pragma unroll
    for (int g = 0; g < 4; ++g) {
      *(v4bf*)&As[buf][aRow[g]][aK[g]] = cvt4(aR[g]);
      v4bf b4 = cvt4(bR[g]);
      Bs[buf][bN4[g] + 0][bKK[g]] = b4.x;
      Bs[buf][bN4[g] + 1][bKK[g]] = b4.y;
      Bs[buf][bN4[g] + 2][bKK[g]] = b4.z;
      Bs[buf][bN4[g] + 3][bKK[g]] = b4.w;
    }
  };

  v8f acc[4][2];
#pragma unroll
  for (int i = 0; i < 4; ++i)
#pragma unroll
    for (int j = 0; j < 2; ++j)
      acc[i][j] = (v8f){0.f, 0.f, 0.f, 0.f, 0.f, 0.f, 0.f, 0.f};

  // ---- prologue: tile 0 into buffer 0 ----
  load_regs(0);
  commit(0);
  __syncthreads();

  const int T = K / BK;
  for (int t = 0; t < T; ++t) {
    const int cur = t & 1;
    // issue next tile's global loads; WMMAs below hide their latency
    if (t + 1 < T) load_regs((t + 1) * BK);

    // ---- fragments (contiguous k order: lane half sel holds k = 16*sel..+15 in A and B) ----
    v16bf afrag[4];
#pragma unroll
    for (int i = 0; i < 4; ++i) {
      int r = wm * 64 + i * 16 + m16;
      v8bf lo = *(const v8bf*)&As[cur][r][sel * 16];        // K = sel*16..+7
      v8bf hi = *(const v8bf*)&As[cur][r][sel * 16 + 8];    // K = sel*16+8..+15
      afrag[i] = __builtin_shufflevector(lo, hi, 0, 1, 2, 3, 4, 5, 6, 7,
                                         8, 9, 10, 11, 12, 13, 14, 15);
    }
    v16bf bfrag[2];
#pragma unroll
    for (int j = 0; j < 2; ++j) {
      int n = wn * 32 + j * 16 + m16;
      bfrag[j] = *(const v16bf*)&Bs[cur][n][sel * 16];  // K = sel*16..+15
    }
#pragma unroll
    for (int i = 0; i < 4; ++i)
#pragma unroll
      for (int j = 0; j < 2; ++j)
        acc[i][j] = __builtin_amdgcn_wmma_f32_16x16x32_bf16(
            false, afrag[i], false, bfrag[j], (short)0, acc[i][j],
            false, false);

    // convert + store next tile into the other buffer (overlaps WMMAs)
    if (t + 1 < T) commit(cur ^ 1);
    __syncthreads();
  }

  // ---- epilogue: C layout lanes 0-15 -> M=r, lanes 16-31 -> M=8+r ----
#pragma unroll
  for (int i = 0; i < 4; ++i) {
#pragma unroll
    for (int j = 0; j < 2; ++j) {
      const int col = col0 + wn * 32 + j * 16 + m16;
      const int rbs = row0 + wm * 64 + i * 16 + sel * 8;
#pragma unroll
      for (int r = 0; r < 8; ++r) {
        const size_t idx = (size_t)(rbs + r) * N + col;
        float val = acc[i][j][r];
        if (Resid) val += Resid[idx];
        C[idx] = val;
      }
    }
  }
}

// ---------------------------------------------------------------------------
// Attention core: block = (head, 64 query rows). k/v head slices staged into
// LDS by the Tensor Data Mover (2D tile 64x512, stride 4096; k uses TDM LDS
// padding for a 68-float row stride => conflict-free score loop). ~284KB of
// the 320KB WGP LDS. f32 softmax, ctx via 4-way partial sums.
// ---------------------------------------------------------------------------
__global__ __launch_bounds__(256) void attention_kernel(
    const float* __restrict__ q, const float* __restrict__ k,
    const float* __restrict__ v, const float* __restrict__ pb,
    const float* __restrict__ mask, float* __restrict__ ctx) {
  extern __shared__ float smem[];
  float* ksh  = smem;                       // [512][68] (padded rows)
  float* vsh  = ksh + 512 * 68;             // [512][64]
  float* qsh  = vsh + 512 * 64;             // [64][64]
  float* sc   = qsh + 64 * 64;              // [512]
  float* red  = sc + 512;                   // [256]
  float* part = red + 256;                  // [4][64]

  const int hd  = blockIdx.x;
  const int q0  = blockIdx.y * 64;
  const int tid = threadIdx.x;
  const int cb  = hd * DK_DIM;

#if HAVE_TDM
  if (tid < 32) {  // one wave owns the two TDM descriptors
    // ---- D# group0/group1 per CDNA5 ISA 8.3/8.4 ----
    const unsigned long long ka = (unsigned long long)(uintptr_t)(k + cb);
    const unsigned long long va = (unsigned long long)(uintptr_t)(v + cb);
    const unsigned kshOff = (unsigned)(uintptr_t)(void*)ksh;  // LDS byte addr
    const unsigned vshOff = (unsigned)(uintptr_t)(void*)vsh;

    // group0: count=1 | lds_addr | global_addr[31:0] | addr[56:32], type=2
    u32x4 g0k = {1u, kshOff, (unsigned)(ka & 0xFFFFFFFFull),
                 (unsigned)((ka >> 32) & 0x01FFFFFFull) | (2u << 30)};
    u32x4 g0v = {1u, vshOff, (unsigned)(va & 0xFFFFFFFFull),
                 (unsigned)((va >> 32) & 0x01FFFFFFull) | (2u << 30)};

    // group1 word0: data_size=2 (4B) [16:17]; k adds pad_enable(bit20),
    // pad_interval=5 (64 DWORDs) [24:22], pad_amount=3 (4 DWORDs) [31:25]
    const int w0k = (int)((2u << 16) | (1u << 20) | (5u << 22) | (3u << 25));
    const int w0v = (int)(2u << 16);
    // tensor_dim0=64 (w1[31:16]), tensor_dim1=512 (w2[31:16]),
    // tile_dim0=64 (w3[31:16]), tile_dim1=512 (w4[15:0]),
    // tensor_dim0_stride=4096 (w5)
    const int w1 = (int)(64u << 16);
    const int w2 = (int)(512u << 16);
    const int w3 = (int)(64u << 16);
    const int w4 = (int)512u;
    const int w5 = (int)4096u;
    i32x8 g1k = {w0k, w1, w2, w3, w4, w5, 0, 0};
    i32x8 g1v = {w0v, w1, w2, w3, w4, w5, 0, 0};
    i32x4 z4 = {0, 0, 0, 0};
#if __clang_major__ >= 23
    i32x8 z8 = {0, 0, 0, 0, 0, 0, 0, 0};
    __builtin_amdgcn_tensor_load_to_lds(g0k, g1k, z4, z4, z8, 0);
    __builtin_amdgcn_tensor_load_to_lds(g0v, g1v, z4, z4, z8, 0);
#else
    __builtin_amdgcn_tensor_load_to_lds(g0k, g1k, z4, z4, 0);
    __builtin_amdgcn_tensor_load_to_lds(g0v, g1v, z4, z4, 0);
#endif
  }
#else
  for (int g = 0; g < 32; ++g) {
    int slot = tid + g * 256;               // 0..8191
    int t  = slot >> 4;
    int d4 = (slot & 15) * 4;
    v4f kv = *(const v4f*)(k + (size_t)t * D_DIM + cb + d4);
    ksh[t * 68 + d4 + 0] = kv.x; ksh[t * 68 + d4 + 1] = kv.y;
    ksh[t * 68 + d4 + 2] = kv.z; ksh[t * 68 + d4 + 3] = kv.w;
    v4f vv = *(const v4f*)(v + (size_t)t * D_DIM + cb + d4);
    vsh[t * 64 + d4 + 0] = vv.x; vsh[t * 64 + d4 + 1] = vv.y;
    vsh[t * 64 + d4 + 2] = vv.z; vsh[t * 64 + d4 + 3] = vv.w;
  }
#endif

  // stage q rows (small; plain loads overlap the TDM transfers)
  for (int g = 0; g < 4; ++g) {
    int slot = tid + g * 256;               // 0..1023
    int r  = slot >> 4;
    int d4 = (slot & 15) * 4;
    v4f qv = *(const v4f*)(q + (size_t)(q0 + r) * D_DIM + cb + d4);
    qsh[r * 64 + d4 + 0] = qv.x; qsh[r * 64 + d4 + 1] = qv.y;
    qsh[r * 64 + d4 + 2] = qv.z; qsh[r * 64 + d4 + 3] = qv.w;
  }
#if HAVE_TDM
  if (tid < 32) __builtin_amdgcn_s_wait_tensorcnt(0);
#endif
  __syncthreads();

  for (int r = 0; r < 64; ++r) {
    const int srow = q0 + r;
    // scores for t = tid and t = tid+256  (no 1/sqrt(d) in T5)
    float s0 = 0.f, s1 = 0.f;
    for (int kk = 0; kk < 64; ++kk) {
      float qv = qsh[r * 64 + kk];
      s0 += qv * ksh[tid * 68 + kk];
      s1 += qv * ksh[(tid + 256) * 68 + kk];
    }
    const float* pbrow = pb + ((size_t)hd * S_LEN + srow) * S_LEN;
    s0 += pbrow[tid] + mask[tid];
    s1 += pbrow[tid + 256] + mask[tid + 256];

    // row max
    red[tid] = fmaxf(s0, s1);
    __syncthreads();
    for (int off = 128; off > 0; off >>= 1) {
      if (tid < off) red[tid] = fmaxf(red[tid], red[tid + off]);
      __syncthreads();
    }
    const float mx = red[0];
    __syncthreads();

    // exp + row sum
    const float e0 = __expf(s0 - mx), e1 = __expf(s1 - mx);
    sc[tid] = e0; sc[tid + 256] = e1;
    red[tid] = e0 + e1;
    __syncthreads();
    for (int off = 128; off > 0; off >>= 1) {
      if (tid < off) red[tid] += red[tid + off];
      __syncthreads();
    }
    const float inv = 1.f / red[0];
    __syncthreads();

    // ctx[srow][dk] = inv * sum_t sc[t] * v[t][dk] with 4-way partials
    const int dk = tid & 63;
    const int p4 = tid >> 6;
    float ps = 0.f;
    for (int t = p4 * 128; t < p4 * 128 + 128; ++t)
      ps += sc[t] * vsh[t * 64 + dk];
    part[p4 * 64 + dk] = ps;
    __syncthreads();
    if (tid < 64) {
      float val = (part[dk] + part[64 + dk] + part[128 + dk] +
                   part[192 + dk]) * inv;
      ctx[(size_t)srow * D_DIM + cb + dk] = val;
    }
    __syncthreads();
  }
}

// ---------------------------------------------------------------------------
// Gated GELU (tanh approx): h = gelu(g0) * g1
// ---------------------------------------------------------------------------
__global__ __launch_bounds__(256) void geglu_kernel(
    const float* __restrict__ g0, const float* __restrict__ g1,
    float* __restrict__ h, int n) {
  int i = blockIdx.x * 256 + threadIdx.x;
  if (i < n) {
    float x = g0[i];
    float t = tanhf(0.7978845608028654f * (x + 0.044715f * x * x * x));
    h[i] = 0.5f * x * (1.f + t) * g1[i];
  }
}

// ---------------------------------------------------------------------------
extern "C" void kernel_launch(void* const* d_in, const int* in_sizes, int n_in,
                              void* d_out, int out_size, void* d_ws,
                              size_t ws_size, hipStream_t stream) {
  (void)in_sizes; (void)n_in; (void)out_size; (void)ws_size;
  const float* x    = (const float*)d_in[0];
  const float* mask = (const float*)d_in[1];
  const float* pb   = (const float*)d_in[2];
  const float* ln1w = (const float*)d_in[3];
  const float* wq   = (const float*)d_in[4];
  const float* wk   = (const float*)d_in[5];
  const float* wv   = (const float*)d_in[6];
  const float* wo   = (const float*)d_in[7];
  const float* ln2w = (const float*)d_in[8];
  const float* wi0  = (const float*)d_in[9];
  const float* wi1  = (const float*)d_in[10];
  const float* wout = (const float*)d_in[11];
  float* out = (float*)d_out;

  const size_t SD = (size_t)S_LEN * D_DIM;   // 2,097,152
  const size_t SF = (size_t)S_LEN * F_DIM;   // 5,242,880
  float* n1  = (float*)d_ws;
  float* qb  = n1  + SD;
  float* kb  = qb  + SD;
  float* vb  = kb  + SD;
  float* ctx = vb  + SD;
  float* xa  = ctx + SD;
  float* n2  = xa  + SD;
  float* g0  = n2  + SD;
  float* g1  = g0  + SF;
  float* hb  = g1  + SF;                      // total ~121.6 MB of d_ws

  const size_t attn_lds =
      (size_t)(512 * 68 + 512 * 64 + 64 * 64 + 512 + 256 + 256) * sizeof(float);
  static bool attr_set = false;  // idempotent host-side attribute, not stream work
  if (!attr_set) {
    (void)hipFuncSetAttribute((const void*)attention_kernel,
                              hipFuncAttributeMaxDynamicSharedMemorySize,
                              (int)attn_lds);
    attr_set = true;
  }

  dim3 gQKV(S_LEN / BM, D_DIM / BN, 3);   // (4, 32, 3) fused q/k/v
  dim3 gD(S_LEN / BM, D_DIM / BN, 1);     // (4, 32)
  dim3 gFF(S_LEN / BM, F_DIM / BN, 2);    // (4, 80, 2) fused wi0/wi1

  // --- self-attention sublayer ---
  rmsnorm_f32_kernel<<<S_LEN, 256, 0, stream>>>(x, ln1w, n1);
  gemm_bf16_wmma_kernel<<<gQKV, 256, 0, stream>>>(
      n1, wq, wk, wv, nullptr, qb, kb, vb, S_LEN, D_DIM, D_DIM);
  attention_kernel<<<dim3(H_NUM, S_LEN / 64), 256, attn_lds, stream>>>(
      qb, kb, vb, pb, mask, ctx);
  gemm_bf16_wmma_kernel<<<gD, 256, 0, stream>>>(
      ctx, wo, wo, wo, x, xa, xa, xa, S_LEN, D_DIM, D_DIM);

  // --- gated FF sublayer ---
  rmsnorm_f32_kernel<<<S_LEN, 256, 0, stream>>>(xa, ln2w, n2);
  gemm_bf16_wmma_kernel<<<gFF, 256, 0, stream>>>(
      n2, wi0, wi1, wi1, nullptr, g0, g1, g1, S_LEN, F_DIM, D_DIM);
  geglu_kernel<<<(int)((SF + 255) / 256), 256, 0, stream>>>(g0, g1, hb,
                                                            (int)SF);
  gemm_bf16_wmma_kernel<<<gD, 256, 0, stream>>>(
      hb, wout, wout, wout, xa, out, out, out, S_LEN, D_DIM, F_DIM);
}
